// AttentionBlock_72885595013390
// MI455X (gfx1250) — matrix operand order repaired and checked by reading, not compiled, not run on hardware
//
#include <hip/hip_runtime.h>
#include <hip/hip_bf16.h>

// ---------------- problem constants ----------------
#define TT    32
#define SS    1024          // H*W
#define CC    768
#define C3    2304          // 3*C
#define HEADS 12
#define HD    64
#define NROW  (TT * SS)     // 32768 rows

typedef __bf16 v16bf __attribute__((ext_vector_type(16)));
typedef float  v8f   __attribute__((ext_vector_type(8)));
typedef unsigned short u16;

union Frag { v16bf v; u16 u[16]; uint4 q[2]; };

static __device__ inline u16 f32_to_bf16(float f) {
  unsigned int u = __float_as_uint(f);
  unsigned int r = (u + 0x7FFFu + ((u >> 16) & 1u)) >> 16;
  return (u16)r;
}
static __device__ inline float bf16_to_f32(u16 s) {
  return __uint_as_float(((unsigned int)s) << 16);
}
static __device__ inline v8f zero8() {
  v8f z;
#pragma unroll
  for (int i = 0; i < 8; ++i) z[i] = 0.f;
  return z;
}
static __device__ inline v8f wmma_bf16(const Frag& a, const Frag& b, v8f c) {
  return __builtin_amdgcn_wmma_f32_16x16x32_bf16(false, a.v, false, b.v,
                                                 (short)0, c, false, false);
}
// A-matrix 16x32 bf16 fragment: lane holds row (lane&15);
// 16 consecutive K at K = (lane>=16)?16:0
static __device__ inline void load_frag_a(Frag& f, const u16* p, int ld,
                                          int row, int k0, int lane) {
  int r = row + (lane & 15);
  int b = (lane >> 4) * 16;
  const u16* q = p + r * ld + k0 + b;
  f.q[0] = *(const uint4*)(q);
  f.q[1] = *(const uint4*)(q + 8);
}
// B-matrix 32x16 bf16 fragment from N-major storage: lane holds col (lane&15),
// 16 consecutive K at K = (lane>=16)?16:0
static __device__ inline void load_frag_b(Frag& f, const u16* p, int ld,
                                          int col, int k0, int lane) {
  int c = col + (lane & 15);
  int b = (lane >> 4) * 16;
  const u16* q = p + c * ld + k0 + b;
  f.q[0] = *(const uint4*)(q);
  f.q[1] = *(const uint4*)(q + 8);
}

// ---------------- weight conversion f32 -> bf16 ----------------
__global__ __launch_bounds__(256) void cvt_w_kernel(const float* __restrict__ wq,
                                                    const float* __restrict__ wo,
                                                    u16* __restrict__ oq,
                                                    u16* __restrict__ oo) {
  const int n1 = CC * C3, n2 = CC * CC;
  int stride = gridDim.x * blockDim.x;
  for (int i = blockIdx.x * blockDim.x + threadIdx.x; i < n1; i += stride)
    oq[i] = f32_to_bf16(wq[i]);
  for (int i = blockIdx.x * blockDim.x + threadIdx.x; i < n2; i += stride)
    oo[i] = f32_to_bf16(wo[i]);
}

// ---------------- InstanceNorm stats: per (frame t, channel c) over 1024 ----
__global__ __launch_bounds__(CC) void stats_kernel(const float* __restrict__ in,
                                                   float* __restrict__ mean,
                                                   float* __restrict__ rstd,
                                                   float eps) {
  int t = blockIdx.x;         // frame
  int c = threadIdx.x;        // channel (768 threads)
  const float* p = in + (size_t)t * SS * CC + c;
  float s = 0.f, s2 = 0.f;
#pragma unroll 4
  for (int i = 0; i < SS; ++i) {
    float v = p[(size_t)i * CC];
    s += v; s2 += v * v;
  }
  float m = s * (1.f / SS);
  float var = s2 * (1.f / SS) - m * m;
  mean[t * CC + c] = m;
  rstd[t * CC + c] = rsqrtf(var + eps);
}

// ---------------- normalize + affine + f32->bf16 ----------------
__global__ __launch_bounds__(256) void norm_kernel(const float* __restrict__ in,
                                                   const float* __restrict__ mean,
                                                   const float* __restrict__ rstd,
                                                   const float* __restrict__ w,
                                                   const float* __restrict__ b,
                                                   u16* __restrict__ out) {
  int r = blockIdx.x;         // row 0..NROW-1
  int t = r >> 10;            // frame
  for (int c = threadIdx.x; c < CC; c += blockDim.x) {
    float v = (in[(size_t)r * CC + c] - mean[t * CC + c]) * rstd[t * CC + c]
              * w[c] + b[c];
    out[(size_t)r * CC + c] = f32_to_bf16(v);
  }
}

// ---------------- WMMA GEMM: (M x K) bf16 @ (K x N) bf16 + bias ------------
// EPI 0: store bf16.  EPI 1: out_f = (acc+bias)*gamma + residual (f32).
template <int EPI>
__global__ __launch_bounds__(256) void gemm_kernel(
    const u16* __restrict__ A, const u16* __restrict__ Bm,
    const float* __restrict__ bias, int M, int K, int Nn,
    u16* __restrict__ out_bf, float* __restrict__ out_f,
    const float* __restrict__ gamma, const float* __restrict__ residual) {
  constexpr int BM = 128, BN = 64, BK = 32;
  __shared__ __align__(16) u16 As[BM][BK];   // 8 KB
  __shared__ __align__(16) u16 Bs[BN][BK];   // 4 KB, N-major (transposed)
  const int bm = blockIdx.y * BM;
  const int bn = blockIdx.x * BN;
  const int tid = threadIdx.x;
  const int lane = tid & 31, wave = tid >> 5;
  const int wm = (wave >> 1) * 32;           // 4 waves along M
  const int wn = (wave & 1) * 32;            // 2 waves along N

  v8f acc[2][2];
#pragma unroll
  for (int i = 0; i < 2; ++i)
#pragma unroll
    for (int j = 0; j < 2; ++j) acc[i][j] = zero8();

  const int ar = tid >> 1, ac = (tid & 1) * 16;   // A-tile loader coords
  const int br = tid >> 3, bc = (tid & 7) * 8;    // B-tile loader coords

  for (int kk = 0; kk < K; kk += BK) {
    // stage A (128x32) — 2x16B per thread, coalesced
    {
      const u16* g = &A[(size_t)(bm + ar) * K + kk + ac];
      *(uint4*)&As[ar][ac]     = *(const uint4*)(g);
      *(uint4*)&As[ar][ac + 8] = *(const uint4*)(g + 8);
      if (kk + BK < K) __builtin_prefetch(g + BK, 0, 1);   // global_prefetch
    }
    // stage B (32x64) transposed to N-major
    {
      uint4 bv = *(const uint4*)&Bm[(size_t)(kk + br) * Nn + bn + bc];
      const u16* bs = (const u16*)&bv;
#pragma unroll
      for (int e = 0; e < 8; ++e) Bs[bc + e][br] = bs[e];
    }
    __syncthreads();

    Frag fa[2], fb[2];
#pragma unroll
    for (int mt = 0; mt < 2; ++mt)
      load_frag_a(fa[mt], &As[0][0], BK, wm + mt * 16, 0, lane);
#pragma unroll
    for (int nt = 0; nt < 2; ++nt)
      load_frag_b(fb[nt], &Bs[0][0], BK, wn + nt * 16, 0, lane);
#pragma unroll
    for (int mt = 0; mt < 2; ++mt)
#pragma unroll
      for (int nt = 0; nt < 2; ++nt)
        acc[mt][nt] = wmma_bf16(fa[mt], fb[nt], acc[mt][nt]);
    __syncthreads();
  }

  // epilogue: D layout — lane holds (row = j + (lane>=16?8:0), col = lane&15)
#pragma unroll
  for (int mt = 0; mt < 2; ++mt)
#pragma unroll
    for (int nt = 0; nt < 2; ++nt) {
      int col = bn + wn + nt * 16 + (lane & 15);
      float bcol = bias[col];
#pragma unroll
      for (int j = 0; j < 8; ++j) {
        int row = bm + wm + mt * 16 + j + ((lane >> 4) << 3);
        float v = acc[mt][nt][j] + bcol;
        if (EPI == 0) {
          out_bf[(size_t)row * Nn + col] = f32_to_bf16(v);
        } else {
          out_f[(size_t)row * Nn + col] =
              v * gamma[col] + residual[(size_t)row * Nn + col];
        }
      }
    }
}

// ---------------- attention: one wave per (spatial n, head) ----------------
static __device__ inline void ln_row_to_lds(const u16* __restrict__ row,
                                            const float* __restrict__ sc,
                                            const float* __restrict__ bi,
                                            u16* __restrict__ dst) {
  float vals[64];
#pragma unroll
  for (int i = 0; i < 8; ++i) {
    uint4 r = *(const uint4*)(row + i * 8);
    const u16* u = (const u16*)&r;
#pragma unroll
    for (int e = 0; e < 8; ++e) vals[i * 8 + e] = bf16_to_f32(u[e]);
  }
  float s = 0.f, s2 = 0.f;
#pragma unroll
  for (int j = 0; j < 64; ++j) { s += vals[j]; s2 += vals[j] * vals[j]; }
  float m = s * (1.f / 64.f);
  float var = s2 * (1.f / 64.f) - m * m;
  float rs = rsqrtf(var + 1e-6f);
#pragma unroll
  for (int j = 0; j < 64; ++j)
    dst[j] = f32_to_bf16((vals[j] - m) * rs * sc[j] + bi[j]);
}

__global__ __launch_bounds__(128) void attn_kernel(
    const u16* __restrict__ qkv,          // [T*S][2304] bf16
    const float* __restrict__ qs, const float* __restrict__ qb,
    const float* __restrict__ ks, const float* __restrict__ kb,
    const float* __restrict__ rel,        // [63][12] f32
    float* __restrict__ outp) {           // [T*S][768] f32
  __shared__ __align__(16) u16 q_lds[4][32][64];   // 16 KB
  __shared__ __align__(16) u16 k_lds[4][32][64];   // 16 KB
  __shared__ __align__(16) u16 p_lds[4][32][32];   //  8 KB
  const int wave = threadIdx.x >> 5, lane = threadIdx.x & 31;
  const int prob = blockIdx.x * 4 + wave;
  const int n = prob / HEADS, head = prob % HEADS;

  // ---- q/k LayerNorm, one time-row per lane, into LDS ----
  {
    int t = lane;
    const u16* base = qkv + (size_t)(t * SS + n) * C3 + head * 192;
    ln_row_to_lds(base,      qs, qb, &q_lds[wave][t][0]);
    ln_row_to_lds(base + 64, ks, kb, &k_lds[wave][t][0]);
  }
  __syncthreads();

  // ---- scores = qn @ kn^T  (M=32 t, N=32 t, K=64 hd) ----
  v8f sc[2][2];
#pragma unroll
  for (int i = 0; i < 2; ++i)
#pragma unroll
    for (int j = 0; j < 2; ++j) sc[i][j] = zero8();
#pragma unroll
  for (int kt = 0; kt < 2; ++kt) {
    Frag fa[2], fb[2];
#pragma unroll
    for (int mt = 0; mt < 2; ++mt)
      load_frag_a(fa[mt], &q_lds[wave][0][0], HD, mt * 16, kt * 32, lane);
#pragma unroll
    for (int nt = 0; nt < 2; ++nt)
      load_frag_b(fb[nt], &k_lds[wave][0][0], HD, nt * 16, kt * 32, lane);
#pragma unroll
    for (int mt = 0; mt < 2; ++mt)
#pragma unroll
      for (int nt = 0; nt < 2; ++nt)
        sc[mt][nt] = wmma_bf16(fa[mt], fb[nt], sc[mt][nt]);
  }

  // ---- scale + relative-position bias ----
#pragma unroll
  for (int mt = 0; mt < 2; ++mt)
#pragma unroll
    for (int nt = 0; nt < 2; ++nt) {
      int col = nt * 16 + (lane & 15);
#pragma unroll
      for (int j = 0; j < 8; ++j) {
        int row = mt * 16 + j + ((lane >> 4) << 3);
        sc[mt][nt][j] = sc[mt][nt][j] * 0.125f +
                        rel[(row - col + (TT - 1)) * HEADS + head];
      }
    }

  // ---- softmax per row (row lives across one 16-lane half) ----
#pragma unroll
  for (int mt = 0; mt < 2; ++mt) {
#pragma unroll
    for (int j = 0; j < 8; ++j) {
      float a0 = sc[mt][0][j], a1 = sc[mt][1][j];
      float m = fmaxf(a0, a1);
      m = fmaxf(m, __shfl_xor(m, 1, 32));
      m = fmaxf(m, __shfl_xor(m, 2, 32));
      m = fmaxf(m, __shfl_xor(m, 4, 32));
      m = fmaxf(m, __shfl_xor(m, 8, 32));
      float e0 = __expf(a0 - m), e1 = __expf(a1 - m);
      float s = e0 + e1;
      s += __shfl_xor(s, 1, 32);
      s += __shfl_xor(s, 2, 32);
      s += __shfl_xor(s, 4, 32);
      s += __shfl_xor(s, 8, 32);
      float inv = 1.f / s;
      int row = mt * 16 + j + ((lane >> 4) << 3);
      p_lds[wave][row][lane & 15]        = f32_to_bf16(e0 * inv);
      p_lds[wave][row][16 + (lane & 15)] = f32_to_bf16(e1 * inv);
    }
  }
  __syncthreads();

  // ---- out = P @ V  (M=32 t, N=64 hd, K=32 t) ----
  Frag pa[2];
#pragma unroll
  for (int mt = 0; mt < 2; ++mt)
    load_frag_a(pa[mt], &p_lds[wave][0][0], 32, mt * 16, 0, lane);

  v8f ov[2][4];
#pragma unroll
  for (int i = 0; i < 2; ++i)
#pragma unroll
    for (int j = 0; j < 4; ++j) ov[i][j] = zero8();

#pragma unroll
  for (int nt = 0; nt < 4; ++nt) {
    Frag fv;
    int hd = nt * 16 + (lane & 15);
    int kb0 = (lane >> 4) * 16;
#pragma unroll
    for (int e = 0; e < 16; ++e) {
      int t2 = kb0 + e;
      fv.u[e] = qkv[(size_t)(t2 * SS + n) * C3 + head * 192 + 128 + hd];
    }
#pragma unroll
    for (int mt = 0; mt < 2; ++mt)
      ov[mt][nt] = wmma_bf16(pa[mt], fv, ov[mt][nt]);
  }

  // ---- scatter to [t*S + n][head*64 + hd] f32 ----
#pragma unroll
  for (int mt = 0; mt < 2; ++mt)
#pragma unroll
    for (int nt = 0; nt < 4; ++nt) {
      int col = nt * 16 + (lane & 15);
#pragma unroll
      for (int j = 0; j < 8; ++j) {
        int t = mt * 16 + j + ((lane >> 4) << 3);
        outp[(size_t)(t * SS + n) * CC + head * HD + col] = ov[mt][nt][j];
      }
    }
}

// ---------------- host side ----------------
static inline size_t al256(size_t x) { return (x + 255) & ~(size_t)255; }

extern "C" void kernel_launch(void* const* d_in, const int* in_sizes, int n_in,
                              void* d_out, int out_size, void* d_ws, size_t ws_size,
                              hipStream_t stream) {
  const float* x        = (const float*)d_in[0];
  const float* norm1_w  = (const float*)d_in[1];
  const float* norm1_b  = (const float*)d_in[2];
  const float* w_qkv    = (const float*)d_in[3];
  const float* b_qkv    = (const float*)d_in[4];
  const float* qnorm_s  = (const float*)d_in[5];
  const float* qnorm_b  = (const float*)d_in[6];
  const float* knorm_s  = (const float*)d_in[7];
  const float* knorm_b  = (const float*)d_in[8];
  const float* rel      = (const float*)d_in[9];
  const float* norm2_w  = (const float*)d_in[10];
  const float* norm2_b  = (const float*)d_in[11];
  const float* w_out    = (const float*)d_in[12];
  const float* b_out    = (const float*)d_in[13];
  const float* gamma    = (const float*)d_in[14];
  float* out = (float*)d_out;

  char* ws = (char*)d_ws;
  size_t off = 0;
  u16* wqkv_bf = (u16*)(ws + off); off += al256((size_t)CC * C3 * 2);
  u16* wout_bf = (u16*)(ws + off); off += al256((size_t)CC * CC * 2);
  float* mean1 = (float*)(ws + off); off += al256((size_t)TT * CC * 4);
  float* rstd1 = (float*)(ws + off); off += al256((size_t)TT * CC * 4);
  float* mean2 = (float*)(ws + off); off += al256((size_t)TT * CC * 4);
  float* rstd2 = (float*)(ws + off); off += al256((size_t)TT * CC * 4);
  u16* xnorm_bf = (u16*)(ws + off); off += al256((size_t)NROW * CC * 2);  // reused
  u16* qkv_bf   = (u16*)(ws + off); off += al256((size_t)NROW * C3 * 2);
  float* attn_f = (float*)(ws + off); off += al256((size_t)NROW * CC * 4);

  // 1. weights -> bf16
  cvt_w_kernel<<<2048, 256, 0, stream>>>(w_qkv, w_out, wqkv_bf, wout_bf);
  // 2. InstanceNorm1 stats + normalize -> bf16
  stats_kernel<<<TT, CC, 0, stream>>>(x, mean1, rstd1, 1e-5f);
  norm_kernel<<<NROW, 256, 0, stream>>>(x, mean1, rstd1, norm1_w, norm1_b, xnorm_bf);
  // 3. qkv GEMM (32768x768)@(768x2304) -> bf16
  gemm_kernel<0><<<dim3(C3 / 64, NROW / 128), 256, 0, stream>>>(
      xnorm_bf, wqkv_bf, b_qkv, NROW, CC, C3,
      qkv_bf, nullptr, nullptr, nullptr);
  // 4. attention over time per (spatial, head)
  attn_kernel<<<(SS * HEADS) / 4, 128, 0, stream>>>(
      qkv_bf, qnorm_s, qnorm_b, knorm_s, knorm_b, rel, attn_f);
  // 5. InstanceNorm2 stats + normalize -> bf16 (reuse xnorm buffer)
  stats_kernel<<<TT, CC, 0, stream>>>(attn_f, mean2, rstd2, 1e-5f);
  norm_kernel<<<NROW, 256, 0, stream>>>(attn_f, mean2, rstd2, norm2_w, norm2_b, xnorm_bf);
  // 6. output GEMM + fused bias, layer-scale gamma, residual
  gemm_kernel<1><<<dim3(CC / 64, NROW / 128), 256, 0, stream>>>(
      xnorm_bf, wout_bf, b_out, NROW, CC, CC,
      nullptr, out, gamma, x);
  (void)in_sizes; (void)n_in; (void)out_size; (void)ws_size;
}
